// QA_CNN_87952340288123
// MI455X (gfx1250) — compile-verified, matrix operand order checked by reading, not run on hardware
//
#include <hip/hip_runtime.h>
#include <hip/hip_bf16.h>

typedef __attribute__((ext_vector_type(16))) __bf16 v16bf;
typedef __attribute__((ext_vector_type(8)))  float  v8f;
typedef __attribute__((ext_vector_type(4)))  float  f32x4;

#define CTXW 5
#define EMB 300
#define NF 100           // real filters
#define CP 112           // filters padded to 7 tiles of 16
#define KP 328           // padded K stride (bf16 elems): 16B aligned, bank-friendly

#if __has_builtin(__builtin_amdgcn_sched_group_barrier)
#define SCHED_GROUP(mask, size, id) __builtin_amdgcn_sched_group_barrier(mask, size, id)
#else
#define SCHED_GROUP(mask, size, id)
#endif
#define SG_WMMA   0x008   // matrix ops
#define SG_DSREAD 0x100   // LDS reads

__device__ __forceinline__ unsigned short f2bf(float f) {
  unsigned int u = __float_as_uint(f);
  u += 0x7FFFu + ((u >> 16) & 1u);    // round-to-nearest-even
  return (unsigned short)(u >> 16);
}

__device__ __forceinline__ v16bf ld_op(const unsigned short* p) {
  union { uint4 q[2]; v16bf v; } u;
  u.q[0] = *(const uint4*)(p);        // K slice +0..7   (VGPR 0-3)
  u.q[1] = *(const uint4*)(p + 16);   // K slice +16..23 (VGPR 4-7)
  return u.v;
}

// Stage transposed bf16 W slice for window j: Wt[c][e] = W[(j*EMB+e)*NF + c]
__device__ __forceinline__ void stage_wt(unsigned short* Wt, const float* __restrict__ W,
                                         int j, int tid, int nt) {
  constexpr int CQ = CP / 4;              // 28 c-quads per e
  for (int i = tid; i < CQ * KP; i += nt) {
    int e  = i / CQ;
    int c4 = (i - e * CQ) * 4;
    f32x4 v = {0.f, 0.f, 0.f, 0.f};
    if (e < EMB && c4 < NF)               // c4 <= 96, quad fully valid
      v = *(const f32x4*)&W[((size_t)(j * EMB + e)) * NF + c4];
    Wt[(c4 + 0) * KP + e] = f2bf(v.x);
    Wt[(c4 + 1) * KP + e] = f2bf(v.y);
    Wt[(c4 + 2) * KP + e] = f2bf(v.z);
    Wt[(c4 + 3) * KP + e] = f2bf(v.w);
  }
}

// out[b,c,l] = sum_{j,e} x[b, l+j-2, e] * W[(j*EMB+e)*NF + c] + bias[c]
// Each wave owns 2 16-l column strips and all 7 c-tiles, so each A-tile (W^T)
// load feeds 2 WMMAs. TWOB=false: strips are adjacent l of one batch
// (block = 256 l). TWOB=true: strips are the same 128 l of two batches.
// Wt is double-buffered: one barrier per window, staging overlaps compute.
template<bool TWOB>
__global__ __launch_bounds__(256) void qa_conv_gemm(
    const float* __restrict__ x, const float* __restrict__ W,
    const float* __restrict__ bias, float* __restrict__ out, int L)
{
  constexpr int NT   = 256;                      // 8 waves
  constexpr int XSEG = TWOB ? (128 + CTXW - 1) : 0;   // rows per batch segment
  constexpr int XR   = TWOB ? 2 * (128 + CTXW - 1) : (256 + CTXW - 1);
  constexpr int QW   = KP / 4;                   // 82 e-quads per staged row

  __shared__ unsigned short Xs[XR * KP];       // bf16 input rows (zero halo + pad)
  __shared__ unsigned short Wt[2][CP * KP];    // double-buffered bf16 W^T slice
  __shared__ float Bs[CP];

  const int tid  = threadIdx.x;
  const int lane = tid & 31;
  const int wave = tid >> 5;
  const int ncol = lane & 15;   // N (l) for B/D, M (c) for A
  const int half = lane >> 4;   // selects K sub-slices per WMMA operand layout
  const int b0   = TWOB ? (2 * blockIdx.y) : blockIdx.y;
  const int l0   = TWOB ? 0 : (blockIdx.x * 256);

  // ---- stage input rows: f32x4 -> 4x bf16 (one b64 ds-store), zero halo/pad ----
  for (int i = tid; i < XR * QW; i += NT) {
    int row = i / QW;
    int e   = (i - row * QW) * 4;
    int seg = TWOB ? (row >= XSEG ? 1 : 0) : 0;
    int r   = TWOB ? (row - seg * XSEG) : row;
    int g   = l0 + r - (CTXW / 2);
    int bb  = b0 + seg;
    f32x4 v = {0.f, 0.f, 0.f, 0.f};
    if (e < EMB && g >= 0 && g < L)       // e <= 296, quad fully valid
      v = __builtin_nontemporal_load(
              (const f32x4*)&x[((size_t)bb * L + g) * EMB + e]);
    union { unsigned short us[4]; uint2 d; } pk;
    pk.us[0] = f2bf(v.x); pk.us[1] = f2bf(v.y);
    pk.us[2] = f2bf(v.z); pk.us[3] = f2bf(v.w);
    *(uint2*)&Xs[row * KP + e] = pk.d;
  }
  stage_wt(Wt[0], W, 0, tid, NT);
  if (tid < CP) Bs[tid] = (tid < NF) ? bias[tid] : 0.f;
  __syncthreads();

  // ---- accumulators, pre-loaded with bias (rows = c, cols = l) ----
  v8f acc[7][2];
  #pragma unroll
  for (int ct = 0; ct < 7; ++ct) {
    #pragma unroll
    for (int r = 0; r < 8; ++r) {
      float bias_v = Bs[ct * 16 + r + 8 * half];
      acc[ct][0][r] = bias_v;
      acc[ct][1][r] = bias_v;
    }
  }

  for (int j = 0; j < CTXW; ++j) {
    // B-operand row base per strip
    const int row0 = TWOB ? (wave * 16 + ncol + j)
                          : (wave * 32 + ncol + j);
    const int row1 = TWOB ? (XSEG + wave * 16 + ncol + j)
                          : (wave * 32 + 16 + ncol + j);
    const unsigned short* xr0 = &Xs[row0 * KP + 8 * half];
    const unsigned short* xr1 = &Xs[row1 * KP + 8 * half];
    const unsigned short* wr  = &Wt[j & 1][ncol * KP + 8 * half];

    #pragma unroll
    for (int t = 0; t < 10; ++t) {          // K = 320 (300 + zero pad)
      const int e0 = t * 32;
      v16bf bv0 = ld_op(xr0 + e0);
      v16bf bv1 = ld_op(xr1 + e0);
      v16bf av[7];
      #pragma unroll
      for (int ct = 0; ct < 7; ++ct)
        av[ct] = ld_op(wr + ct * (16 * KP) + e0);
      #pragma unroll
      for (int ct = 0; ct < 7; ++ct) {
        acc[ct][0] = __builtin_amdgcn_wmma_f32_16x16x32_bf16(
            false, av[ct], false, bv0, (short)0, acc[ct][0], false, false);
        acc[ct][1] = __builtin_amdgcn_wmma_f32_16x16x32_bf16(
            false, av[ct], false, bv1, (short)0, acc[ct][1], false, false);
      }
      // Pattern this k-step: all 18 ds_loads first, then 14 WMMAs -> partial
      // dscnt thresholds, one LDS latency amortized over 14 matrix ops.
      SCHED_GROUP(SG_DSREAD, 18, 0);
      SCHED_GROUP(SG_WMMA,   14, 0);
    }

    if (j < CTXW - 1) {
      // Stage next window into the other buffer; waves overlap this with
      // other waves' compute tails. Single barrier per window.
      stage_wt(Wt[(j + 1) & 1], W, j + 1, tid, NT);
      __syncthreads();
    }
  }

  // ---- epilogue: D rows are c (VGPR r -> c = ct*16 + r + 8*half), cols are l ----
  #pragma unroll
  for (int s = 0; s < 2; ++s) {
    const int bb = TWOB ? (b0 + s) : b0;
    const int l  = TWOB ? (wave * 16 + ncol)
                        : (l0 + wave * 32 + s * 16 + ncol);
    #pragma unroll
    for (int ct = 0; ct < 7; ++ct) {
      #pragma unroll
      for (int r = 0; r < 8; ++r) {
        int c = ct * 16 + r + 8 * half;
        if (c < NF)
          __builtin_nontemporal_store(acc[ct][s][r],
                                      &out[((size_t)bb * NF + c) * L + l]);
      }
    }
  }
}

extern "C" void kernel_launch(void* const* d_in, const int* in_sizes, int n_in,
                              void* d_out, int out_size, void* d_ws, size_t ws_size,
                              hipStream_t stream) {
  const float* question = (const float*)d_in[0];   // [B,128,300] f32
  const float* answer   = (const float*)d_in[1];   // [B,512,300] f32
  const float* W        = (const float*)d_in[2];   // [1500,100]  f32
  const float* bias     = (const float*)d_in[3];   // [100]       f32

  const int Mq = 128, La = 512;
  const int B = in_sizes[0] / (Mq * EMB);          // 256

  float* outQ = (float*)d_out;                     // [B,100,128]
  float* outA = outQ + (size_t)B * NF * Mq;        // [B,100,512]

  // question: strips = 2 batches x 128 l, LDS ~313 KB (double-buffered Wt)
  qa_conv_gemm<true><<<dim3(1, B / 2), 256, 0, stream>>>(question, W, bias, outQ, Mq);
  // answer: strips = 2 adjacent 16-l columns (256 l/block), LDS ~310.5 KB
  qa_conv_gemm<false><<<dim3(La / 256, B), 256, 0, stream>>>(answer, W, bias, outA, La);
}
